// Encoder_55757265436854
// MI455X (gfx1250) — matrix operand order repaired and checked by reading, not compiled, not run on hardware
//
#include <hip/hip_runtime.h>

typedef _Float16 h16;
typedef __attribute__((ext_vector_type(16))) _Float16 v16h;
typedef __attribute__((ext_vector_type(8)))  _Float16 v8h;
typedef __attribute__((ext_vector_type(8)))  float    v8f;

#define NN    2048
#define ESP   65536
#define CC    128
#define LROW  40          // padded LDS row stride (halfs): 16B aligned, conflict-free

__device__ __forceinline__ float sigmoidf_(float x) {
    return 1.0f / (1.0f + expf(-x));
}

// ---------------- degree / dinv ----------------
__global__ void k_deg_init(float* deg) {
    int i = blockIdx.x * blockDim.x + threadIdx.x;
    if (i < NN) deg[i] = 1.0f;                       // self-loop weight
}

__global__ void k_deg_sparse(const int* __restrict__ ei, float* deg) {
    int e = blockIdx.x * blockDim.x + threadIdx.x;
    if (e < ESP) atomicAdd(&deg[ei[ESP + e]], 1.0f); // dst = ei[1][e]
}

// deg[c] += sum_r sigmoid(masked_y[r, c]) for c < 1024 (coalesced over c)
__global__ void k_deg_dense(const float* __restrict__ my, float* deg) {
    int b  = blockIdx.x;                 // 32 blocks
    int c  = (b & 3) * 256 + threadIdx.x;
    int r0 = (b >> 2) * 128;
    float s = 0.0f;
    for (int r = r0; r < r0 + 128; ++r)
        s += sigmoidf_(my[(size_t)r * NN + c]);
    atomicAdd(&deg[c], s);
}

__global__ void k_dinv(const float* __restrict__ deg, float* dinv) {
    int i = blockIdx.x * blockDim.x + threadIdx.x;
    if (i < NN) { float d = deg[i]; dinv[i] = d > 0.0f ? rsqrtf(d) : 0.0f; }
}

// ---------------- build A_hat (f32 staging) ----------------
__global__ void k_dense_block(const float* __restrict__ my,
                              const float* __restrict__ dinv, float* A32) {
    int idx = blockIdx.x * blockDim.x + threadIdx.x;   // 1M threads
    int d = idx >> 10, s = idx & 1023;
    float w = sigmoidf_(my[(size_t)s * NN + d]);       // edge s->d, weight sig(my[s,d])
    A32[(size_t)d * NN + s] = dinv[d] * dinv[s] * w;
}

__global__ void k_diag(const float* __restrict__ dinv, float* A32) {
    int i = blockIdx.x * blockDim.x + threadIdx.x;
    if (i < NN) A32[(size_t)i * NN + i] += dinv[i] * dinv[i];
}

__global__ void k_scatter(const int* __restrict__ ei,
                          const float* __restrict__ dinv, float* A32) {
    int e = blockIdx.x * blockDim.x + threadIdx.x;
    if (e < ESP) {
        int s = ei[e], d = ei[ESP + e];
        atomicAdd(&A32[(size_t)d * NN + s], dinv[d] * dinv[s]);
    }
}

__global__ void k_convert(const float* __restrict__ A32, h16* __restrict__ A16) {
    int i = (blockIdx.x * blockDim.x + threadIdx.x) * 8;
    v8h o;
    #pragma unroll
    for (int j = 0; j < 8; ++j) o[j] = (h16)A32[i + j];
    *(v8h*)(A16 + i) = o;
}

// ---------------- prep: f16 inputs / transposed weights ----------------
__global__ void k_x16(const float* __restrict__ x, h16* __restrict__ x16) {
    int i = blockIdx.x * blockDim.x + threadIdx.x;
    if (i < NN * CC) x16[i] = (h16)x[i];
}

__global__ void k_prep_w(const float* __restrict__ W1,
                         const float* __restrict__ Wmu, const float* __restrict__ Wls,
                         const float* __restrict__ bmu, const float* __restrict__ bls,
                         h16* __restrict__ W1t, h16* __restrict__ Wct,
                         float* __restrict__ bcat) {
    int i = blockIdx.x * blockDim.x + threadIdx.x;  // 16384
    if (i < CC * CC) {
        int n = i >> 7, k = i & 127;
        W1t[i] = (h16)W1[k * CC + n];                      // W1t[n][k]
        Wct[i] = (n < 64) ? (h16)Wmu[k * 64 + n]
                          : (h16)Wls[k * 64 + (n - 64)];   // [Wmu|Wls]^T
        if (i < CC) bcat[i] = (i < 64) ? bmu[i] : bls[i - 64];
    }
}

// ---------------- WMMA GEMM with LDS-staged B + split-K ----------------
// A: f16 row-major (lda). Bt: f16 transposed, Bt[n*ldbt + k].
// Block = 8 waves, each wave: 16 rows x 128 cols. Grid: (M/128, KSPLIT).
// Kc = K / KSPLIT columns of K per block.
// If Cs != nullptr: write f32 partials to Cs[kz][m][n] (deterministic split-K).
// Else (single split): store f16 transposed outT[n*2048 + m].
__global__ __launch_bounds__(256)
void k_gemm(const h16* __restrict__ A, int lda,
            const h16* __restrict__ Bt, int ldbt,
            int Kc, int Ktot,
            h16* __restrict__ outT, float* __restrict__ Cs) {
    __shared__ __align__(32) h16 lb[2][128 * LROW];   // double-buffered B k-slice

    const int tid  = threadIdx.x;
    const int lane = tid & 31;
    const int wave = tid >> 5;
    const int g    = lane >> 4;
    const int ln   = lane & 15;
    const int m0   = blockIdx.x * 128 + wave * 16;
    const int kbeg = blockIdx.y * Kc;
    const int kend = kbeg + Kc;

    const h16* Arow = A + (size_t)(m0 + ln) * lda;

    // B-slice staging indices: thread t covers row n = t/2, 32B half-row h = t&1
    const int snr = tid >> 1;
    const int shh = (tid & 1) * 16;

    v8f acc[8];
    v8f vz = {};
    #pragma unroll
    for (int i = 0; i < 8; ++i) acc[i] = vz;

    // prologue: stage first slice
    {
        v16h v = *(const v16h*)(Bt + (size_t)snr * ldbt + kbeg + shh);
        *(v16h*)(&lb[0][snr * LROW + shh]) = v;
    }
    __syncthreads();

    const int nk = Kc >> 5;
    for (int it = 0; it < nk; ++it) {
        const int kt  = kbeg + (it << 5);
        const int cur = it & 1;

        // A fragment: element e -> K=kt+16g+e
        v16h af = *(const v16h*)(Arow + kt + 16 * g);

        if (kt + 288 <= kend)
            __builtin_prefetch((const void*)(Arow + kt + 256), 0, 3);

        // all 8 B fragments from LDS (batched -> waits grouped, wmma back-to-back)
        v16h bf[8];
        #pragma unroll
        for (int nt = 0; nt < 8; ++nt)
            bf[nt] = *(const v16h*)(&lb[cur][(nt * 16 + ln) * LROW + 16 * g]);

        __syncthreads();                       // LDS reads done -> safe to overwrite

        if (it + 1 < nk) {                     // stage next slice into other buffer
            v16h v = *(const v16h*)(Bt + (size_t)snr * ldbt + (kt + 32) + shh);
            *(v16h*)(&lb[cur ^ 1][snr * LROW + shh]) = v;
        }

        #pragma unroll
        for (int nt = 0; nt < 8; ++nt)
            acc[nt] = __builtin_amdgcn_wmma_f32_16x16x32_f16(
                false, af, false, bf[nt], (short)0, acc[nt], false, false);

        __syncthreads();                       // next slice visible
    }

    if (Cs) {
        float* c = Cs + (size_t)blockIdx.y * NN * CC;
        #pragma unroll
        for (int nt = 0; nt < 8; ++nt) {
            int n = nt * 16 + ln;
            #pragma unroll
            for (int r = 0; r < 8; ++r) {
                int m = m0 + r + 8 * g;
                c[(size_t)m * CC + n] = acc[nt][r];
            }
        }
    } else {
        #pragma unroll
        for (int nt = 0; nt < 8; ++nt) {
            int n = nt * 16 + ln;
            #pragma unroll
            for (int r = 0; r < 8; ++r) {
                int m = m0 + r + 8 * g;
                outT[(size_t)n * NN + m] = (h16)acc[nt][r];
            }
        }
    }
}

// ---------------- split-K reduction + epilogue ----------------
// mode 1: hid[m*128+n] = relu(sum + bias[n])  (f16, row-major)
// mode 2: split f32 store: n<64 -> z_mu[m*64+n] ; else z_ls[m*64+n-64]
__global__ void k_epi(const float* __restrict__ Cs, int KS,
                      const float* __restrict__ bias,
                      h16* __restrict__ out16, float* __restrict__ out32, int mode) {
    int i = blockIdx.x * blockDim.x + threadIdx.x;   // 262144
    int m = i >> 7, n = i & 127;
    float v = 0.0f;
    for (int z = 0; z < KS; ++z) v += Cs[(size_t)z * (NN * CC) + i];
    v += bias[n];
    if (mode == 1) {
        v = v > 0.0f ? v : 0.0f;
        out16[i] = (h16)v;
    } else {
        if (n < 64) out32[(size_t)m * 64 + n] = v;
        else        out32[(size_t)NN * 64 + (size_t)m * 64 + (n - 64)] = v;
    }
}

extern "C" void kernel_launch(void* const* d_in, const int* in_sizes, int n_in,
                              void* d_out, int out_size, void* d_ws, size_t ws_size,
                              hipStream_t stream) {
    const float* x   = (const float*)d_in[0];
    const float* my  = (const float*)d_in[1];
    const float* W1  = (const float*)d_in[2];
    const float* b1  = (const float*)d_in[3];
    const float* Wmu = (const float*)d_in[4];
    const float* bmu = (const float*)d_in[5];
    const float* Wls = (const float*)d_in[6];
    const float* bls = (const float*)d_in[7];
    const int*   ei  = (const int*)d_in[8];
    float* out = (float*)d_out;

    char* w = (char*)d_ws;
    size_t off = 0;
    float* deg  = (float*)(w + off); off += (size_t)NN * 4;
    float* dinv = (float*)(w + off); off += (size_t)NN * 4;
    float* A32  = (float*)(w + off); off += (size_t)NN * NN * 4;     // 16 MB
    h16*   A16  = (h16*)(w + off);   off += (size_t)NN * NN * 2;     // 8 MB
    float* Cs   = (float*)(w + off); off += (size_t)8 * NN * CC * 4; // 8 MB split-K
    h16*   x16  = (h16*)(w + off);   off += (size_t)NN * CC * 2;
    h16*   W1t  = (h16*)(w + off);   off += (size_t)CC * CC * 2;
    h16*   Wct  = (h16*)(w + off);   off += (size_t)CC * CC * 2;
    float* bcat = (float*)(w + off); off += 512;
    h16*   h1T  = (h16*)(w + off);   off += (size_t)CC * NN * 2;
    h16*   hid  = (h16*)(w + off);   off += (size_t)NN * CC * 2;
    h16*   h2T  = (h16*)(w + off);   off += (size_t)CC * NN * 2;

    hipMemsetAsync(A32, 0, (size_t)NN * NN * 4, stream);

    k_deg_init  <<<8,    256, 0, stream>>>(deg);
    k_deg_sparse<<<256,  256, 0, stream>>>(ei, deg);
    k_deg_dense <<<32,   256, 0, stream>>>(my, deg);
    k_dinv      <<<8,    256, 0, stream>>>(deg, dinv);

    k_dense_block<<<4096, 256, 0, stream>>>(my, dinv, A32);
    k_diag       <<<8,    256, 0, stream>>>(dinv, A32);
    k_scatter    <<<256,  256, 0, stream>>>(ei, dinv, A32);
    k_convert    <<<2048, 256, 0, stream>>>(A32, A16);

    k_x16   <<<1024, 256, 0, stream>>>(x, x16);
    k_prep_w<<<64,   256, 0, stream>>>(W1, Wmu, Wls, bmu, bls, W1t, Wct, bcat);

    // G1: h1T = (x @ W1)^T                       [2048 x 128 x 128], 1 split
    k_gemm<<<dim3(16, 1), 256, 0, stream>>>(x16, CC, W1t, CC, CC, CC, h1T, nullptr);
    // G2: hidden = relu(A_hat @ h1 + b1)         [2048 x 2048 x 128], 8 splits
    k_gemm<<<dim3(16, 8), 256, 0, stream>>>(A16, NN, h1T, NN, NN / 8, NN, nullptr, Cs);
    k_epi <<<1024, 256, 0, stream>>>(Cs, 8, b1, hid, nullptr, 1);
    // G3: h2T = (hidden @ [Wmu|Wls])^T           [2048 x 128 x 128], 1 split
    k_gemm<<<dim3(16, 1), 256, 0, stream>>>(hid, CC, Wct, CC, CC, CC, h2T, nullptr);
    // G4: z_mu|z_ls = A_hat @ h2 + [bmu|bls]     [2048 x 2048 x 128], 8 splits
    k_gemm<<<dim3(16, 8), 256, 0, stream>>>(A16, NN, h2T, NN, NN / 8, NN, nullptr, Cs);
    k_epi <<<1024, 256, 0, stream>>>(Cs, 8, bcat, nullptr, out, 2);

    (void)in_sizes; (void)n_in; (void)out_size; (void)ws_size;
}
